// GCN_57397942943830
// MI455X (gfx1250) — matrix operand order repaired and checked by reading, not compiled, not run on hardware
//
#include <hip/hip_runtime.h>

// Problem constants (from reference)
constexpr int BATCH = 8;
constexpr int NODES = 4096;
constexpr int EDGES = 131072;
constexpr int CHAN  = 256;     // C_IN == C_OUT == 256

typedef __attribute__((ext_vector_type(16))) __bf16 v16bf;
typedef __attribute__((ext_vector_type(8)))  __bf16 bf16x8;
typedef __attribute__((ext_vector_type(8)))  float  v8f;

__device__ __forceinline__ __bf16 f2bf(float f) {
    union { float f; unsigned u; } v; v.f = f;
    unsigned r = v.u + 0x7FFFu + ((v.u >> 16) & 1u);   // round-to-nearest-even
    unsigned short h = (unsigned short)(r >> 16);
    return __builtin_bit_cast(__bf16, h);
}

// ---------------------------------------------------------------------------
// GEMM: H[N,256] = X[N,256] @ W[256,256]  (bf16 WMMA, f32 accumulate)
// grid (N/128, 256/32, BATCH), block 256 (8 waves). Wave w owns a 16x32 slab.
// ---------------------------------------------------------------------------
#define TM 128
#define TN 32
#define TK 32
#define LDK 48   // padded row length (96B, 16B-aligned, spreads LDS banks)

__global__ __launch_bounds__(256)
void gcn_gemm_bf16(const float* __restrict__ X, const float* __restrict__ Wm,
                   float* __restrict__ H) {
    __shared__ __align__(16) __bf16 As[TM][LDK];   // 12 KB
    __shared__ __align__(16) __bf16 Bs[TN][LDK];   //  3 KB, stored [n][k]

    const int tid  = threadIdx.x;
    const int wave = tid >> 5;
    const int lane = tid & 31;
    const long gbase = (long)blockIdx.z * NODES * CHAN;
    const int  row0  = blockIdx.x * TM;
    const int  col0  = blockIdx.y * TN;

    v8f c0 = {}; v8f c1 = {};

    for (int ks = 0; ks < CHAN; ks += TK) {
        // stage A tile 128x32 (f32 -> bf16): 1024 float4, 4 per thread
        #pragma unroll
        for (int i = 0; i < 4; ++i) {
            int fid = tid + i * 256;
            int r   = fid >> 3;
            int c4  = (fid & 7) << 2;
            const float4 v = *(const float4*)(X + gbase + (long)(row0 + r) * CHAN + ks + c4);
            As[r][c4 + 0] = f2bf(v.x); As[r][c4 + 1] = f2bf(v.y);
            As[r][c4 + 2] = f2bf(v.z); As[r][c4 + 3] = f2bf(v.w);
        }
        // stage B tile 32x32 transposed -> Bs[n][k]
        {
            int k  = tid >> 3;
            int n4 = (tid & 7) << 2;
            const float4 v = *(const float4*)(Wm + (long)(ks + k) * CHAN + col0 + n4);
            Bs[n4 + 0][k] = f2bf(v.x); Bs[n4 + 1][k] = f2bf(v.y);
            Bs[n4 + 2][k] = f2bf(v.z); Bs[n4 + 3][k] = f2bf(v.w);
        }
        __syncthreads();

        // Build fragments in contiguous K order (k = 16*h + s), same for A and B.
        const int m  = lane & 15;
        const int kg = (lane >> 4) << 4;   // A: K half 0-15 / 16-31
        const int kb = (lane >> 4) << 4;   // B: K half 0-15 / 16-31
        bf16x8 alo  = *(const bf16x8*)&As[wave * 16 + m][kg];
        bf16x8 ahi  = *(const bf16x8*)&As[wave * 16 + m][kg + 8];
        bf16x8 b0lo = *(const bf16x8*)&Bs[m][kb];
        bf16x8 b0hi = *(const bf16x8*)&Bs[m][kb + 8];
        bf16x8 b1lo = *(const bf16x8*)&Bs[m + 16][kb];
        bf16x8 b1hi = *(const bf16x8*)&Bs[m + 16][kb + 8];
        v16bf a, b0, b1;
        #pragma unroll
        for (int i = 0; i < 8; ++i) {
            a[i]  = alo[i];  a[i + 8]  = ahi[i];
            b0[i] = b0lo[i]; b0[i + 8] = b0hi[i];
            b1[i] = b1lo[i]; b1[i + 8] = b1hi[i];
        }
        c0 = __builtin_amdgcn_wmma_f32_16x16x32_bf16(false, a, false, b0, (short)0, c0, false, false);
        c1 = __builtin_amdgcn_wmma_f32_16x16x32_bf16(false, a, false, b1, (short)0, c1, false, false);
        __syncthreads();
    }

    // D layout: VGPR r, lane<16 -> (M=r, N=lane); lane>=16 -> (M=r+8, N=lane-16)
    const int n     = lane & 15;
    const int mhalf = (lane >> 4) << 3;
    float* Hp = H + gbase + (long)(row0 + wave * 16 + mhalf) * CHAN + col0;
    #pragma unroll
    for (int r = 0; r < 8; ++r) {
        Hp[(long)r * CHAN + n]      = c0[r];
        Hp[(long)r * CHAN + 16 + n] = c1[r];
    }
}

// ---------------------------------------------------------------------------
// Degree / normalization helpers
// ---------------------------------------------------------------------------
__global__ void gcn_fill(float* __restrict__ p, float v, long n) {
    long i = (long)blockIdx.x * blockDim.x + threadIdx.x;
    long s = (long)gridDim.x * blockDim.x;
    for (; i < n; i += s) p[i] = v;
}

__global__ void gcn_fill_bias(float* __restrict__ p, const float* __restrict__ bias, long n) {
    long i = (long)blockIdx.x * blockDim.x + threadIdx.x;
    long s = (long)gridDim.x * blockDim.x;
    for (; i < n; i += s) p[i] = bias[i & (CHAN - 1)];
}

__global__ void gcn_deg(const int* __restrict__ ei, float* __restrict__ deg) {
    long i = (long)blockIdx.x * blockDim.x + threadIdx.x;
    if (i >= (long)BATCH * EDGES) return;
    int b = (int)(i / EDGES);
    int e = (int)(i % EDGES);
    int dst = ei[(long)b * 2 * EDGES + EDGES + e];
    unsafeAtomicAdd(&deg[(long)b * NODES + dst], 1.0f);
}

__global__ void gcn_rsqrt(float* __restrict__ p, long n) {
    long i = (long)blockIdx.x * blockDim.x + threadIdx.x;
    if (i < n) { float d = p[i]; p[i] = (d > 0.0f) ? rsqrtf(d) : 0.0f; }
}

__global__ void gcn_relu(float* __restrict__ p, long n) {
    long i = (long)blockIdx.x * blockDim.x + threadIdx.x;
    if (i < n) p[i] = fmaxf(p[i], 0.0f);
}

// ---------------------------------------------------------------------------
// Edge scatter: one wave per (edge or self-loop), 8 coalesced f32 atomics/lane
// ---------------------------------------------------------------------------
__global__ __launch_bounds__(256)
void gcn_scatter(const int* __restrict__ ei, const float* __restrict__ dinv,
                 const float* __restrict__ Hsrc, float* __restrict__ Out) {
    const int wave = threadIdx.x >> 5;
    const int lane = threadIdx.x & 31;
    const long eidx = (long)blockIdx.x * 8 + wave;
    constexpr long PER_G = (long)EDGES + NODES;
    if (eidx >= (long)BATCH * PER_G) return;
    const int b = (int)(eidx / PER_G);
    const int e = (int)(eidx % PER_G);

    const float* dv = dinv + (long)b * NODES;
    int src, dst; float norm;
    if (e < EDGES) {
        const int* eb = ei + (long)b * 2 * EDGES;
        src = eb[e];
        dst = eb[EDGES + e];
        norm = dv[src] * dv[dst];
    } else {            // self loop
        src = dst = e - EDGES;
        float d = dv[src];
        norm = d * d;
    }
    const float* hs = Hsrc + ((long)b * NODES + src) * CHAN;
    float*       op = Out  + ((long)b * NODES + dst) * CHAN;
    #pragma unroll
    for (int j = 0; j < 8; ++j) {
        int c = lane + j * 32;
        unsafeAtomicAdd(&op[c], hs[c] * norm);
    }
}

// ---------------------------------------------------------------------------
extern "C" void kernel_launch(void* const* d_in, const int* in_sizes, int n_in,
                              void* d_out, int out_size, void* d_ws, size_t ws_size,
                              hipStream_t stream) {
    const float* x  = (const float*)d_in[0];
    const int*   ei = (const int*)d_in[1];
    const float* W1 = (const float*)d_in[2];
    const float* b1 = (const float*)d_in[3];
    const float* W2 = (const float*)d_in[4];
    const float* b2 = (const float*)d_in[5];
    float* out = (float*)d_out;

    char* ws = (char*)d_ws;
    float* dinv = (float*)ws;                                  // B*N f32 (128 KB)
    size_t off  = (((size_t)BATCH * NODES * 4) + 255) & ~(size_t)255;
    float* h    = (float*)(ws + off);                          // B*N*256 f32 (32 MB)

    const long nFeat  = (long)BATCH * NODES * CHAN;            // 8.39M
    const long nNode  = (long)BATCH * NODES;                   // 32768
    const long nEdgeT = (long)BATCH * ((long)EDGES + NODES);   // edges + self loops

    dim3 blk(256);
    dim3 gemmGrid(NODES / TM, CHAN / TN, BATCH);               // (32, 8, 8)
    int scatterBlocks = (int)((nEdgeT + 7) / 8);

    // --- degree / norm ---
    gcn_fill<<<512, blk, 0, stream>>>(dinv, 1.0f, nNode);      // self-loop count
    gcn_deg<<<(int)(((long)BATCH * EDGES + 255) / 256), blk, 0, stream>>>(ei, dinv);
    gcn_rsqrt<<<(int)((nNode + 255) / 256), blk, 0, stream>>>(dinv, nNode);

    // --- layer 1 ---
    gcn_gemm_bf16<<<gemmGrid, blk, 0, stream>>>(x, W1, h);
    gcn_fill_bias<<<4096, blk, 0, stream>>>(out, b1, nFeat);
    gcn_scatter<<<scatterBlocks, blk, 0, stream>>>(ei, dinv, h, out);
    gcn_relu<<<(int)((nFeat + 255) / 256), blk, 0, stream>>>(out, nFeat);

    // --- layer 2 ---
    gcn_gemm_bf16<<<gemmGrid, blk, 0, stream>>>(out, W2, h);
    gcn_fill_bias<<<4096, blk, 0, stream>>>(out, b2, nFeat);
    gcn_scatter<<<scatterBlocks, blk, 0, stream>>>(ei, dinv, h, out);
}
